// RelativeMultiHeadAttention_25598005084329
// MI455X (gfx1250) — matrix operand order repaired and checked by reading, not compiled, not run on hardware
//
#include <hip/hip_runtime.h>
#include <math.h>

typedef _Float16 v8h  __attribute__((ext_vector_type(8)));
typedef _Float16 v16h __attribute__((ext_vector_type(16)));
typedef float    v8f  __attribute__((ext_vector_type(8)));

namespace {
constexpr int kB = 2;
constexpr int kL = 2048;   // query length == key length
constexpr int kH = 16;
constexpr int kD = 64;
constexpr int kT = 33;     // 2*MAX_REL+1 rel-table rows
constexpr int kBand = 31;  // |s-l| <= 15 diagonal band
constexpr int kBr = 64;    // query rows per block (4 waves x 16)
constexpr int kBc = 32;    // keys per S-step
constexpr int kThreads = 128;
constexpr int kPF = (kBc * kD) / kThreads;  // 16 prefetch elems/thread (fallback)
constexpr size_t kHeadElems = (size_t)kB * kH * kL * kD;
constexpr size_t kWsNeed = 2 * kHeadElems * sizeof(_Float16);   // Kh + Vt, 16 MB
}

__device__ __forceinline__ v16h cat8(v8h a, v8h b) {
  return __builtin_shufflevector(a, b, 0,1,2,3,4,5,6,7,8,9,10,11,12,13,14,15);
}

// LDS byte offset of a __shared__ pointer (flat LDS addr truncates to low 32 bits)
__device__ __forceinline__ unsigned lds_off(const void* p) {
  return (unsigned)(uintptr_t)p;
}
// CDNA5 async global->LDS 16B copy (ASYNCcnt-tracked)
__device__ __forceinline__ void async_b128(unsigned ldsaddr, const _Float16* g) {
  asm volatile("global_load_async_to_lds_b128 %0, %1, off"
               :: "v"(ldsaddr), "v"(g) : "memory");
}
__device__ __forceinline__ void wait_async0() {
  asm volatile("s_wait_asynccnt 0" ::: "memory");
}

// ======================= pre-pass: f32 K/V -> f16 Kh (head-major) + Vt (transposed)
__global__ __launch_bounds__(256)
void preconvert_kv(const float* __restrict__ K, const float* __restrict__ V,
                   _Float16* __restrict__ Kh, _Float16* __restrict__ Vt) {
  __shared__ _Float16 tile[64][65];   // pad to dodge bank conflicts on transposed read
  const int tid = threadIdx.x;
  const int S0 = blockIdx.x * 64;
  const int h = blockIdx.y, b = blockIdx.z;
  const size_t hb = (size_t)(b * kH + h);
  for (int e = tid; e < 64 * 64; e += 256) {
    int sr = e >> 6, d = e & 63;
    size_t gs = ((size_t)((b * kL + S0 + sr) * kH + h)) * kD + d;
    _Float16 kv = (_Float16)K[gs];
    Kh[(hb * kL + S0 + sr) * kD + d] = kv;
    tile[sr][d] = (_Float16)V[gs];
  }
  __syncthreads();
  for (int e = tid; e < 64 * 64; e += 256) {
    int d = e >> 6, sr = e & 63;
    Vt[(hb * kD + d) * kL + S0 + sr] = tile[sr][d];
  }
}

// ======================= fast path: async-DMA staged flash attention =======================
__global__ __launch_bounds__(kThreads)
void relattn_async(const float* __restrict__ Q, const _Float16* __restrict__ Kh,
                   const _Float16* __restrict__ Vt, const float* __restrict__ RK,
                   const float* __restrict__ RV, float* __restrict__ Out) {
  __shared__ _Float16 sK[2][kBc][kD];     // key tile, row-major f16 (double buffered)
  __shared__ _Float16 sVt[2][kD][kBc];    // value tile, d-major f16 (double buffered)
  __shared__ float    sQR[kBr][kT];
  __shared__ float    sBand[kBr][kBand];
  __shared__ _Float16 sP[4][16][kBc];
  __shared__ float    sO[kBr][kD];
  __shared__ float    sStat[kBr][3];

  const int tid  = threadIdx.x;
  const int wave = tid >> 5;
  const int lane = tid & 31;
  const int ln   = lane & 15;
  const int lh   = lane >> 4;

  const int L0 = blockIdx.x * kBr;
  const int h  = blockIdx.y;
  const int b  = blockIdx.z;
  const _Float16* KhB = Kh + (size_t)(b * kH + h) * kL * kD;
  const _Float16* VtB = Vt + (size_t)(b * kH + h) * kD * kL;

  // ---- issue async copies for tile starting at s0n into buffer bufn ----
  auto issueTile = [&](int s0n, int bufn) {
    #pragma unroll
    for (int i = 0; i < 2; ++i) {
      int c = tid + i * kThreads;            // 256 x 16B chunks per array
      int sr = c >> 3, d8 = (c & 7) * 8;     // K: 32 rows x 8 chunks
      async_b128(lds_off(&sK[bufn][sr][d8]), KhB + ((size_t)(s0n + sr) * kD + d8));
      int dv = c >> 2, s8 = (c & 3) * 8;     // Vt: 64 rows x 4 chunks
      async_b128(lds_off(&sVt[bufn][dv][s8]), VtB + ((size_t)dv * kL + s0n + s8));
    }
  };

  // ---- precompute qrk table (f32) and init band logits ----
  for (int e = tid; e < kBr * kT; e += kThreads) {
    int m = e / kT, t = e - m * kT;
    const float* qrow = Q + ((size_t)((b * kL + L0 + m) * kH + h)) * kD;
    const float* rkrow = RK + t * kD;
    float s = 0.f;
    #pragma unroll
    for (int d = 0; d < kD; ++d) s += qrow[d] * rkrow[d];
    sQR[m][t] = s;
  }
  for (int e = tid; e < kBr * kBand; e += kThreads)
    sBand[e / kBand][e % kBand] = -INFINITY;

  // ---- preload this lane's Q A-fragments ----
  const int l0w = L0 + wave * 16;
  const float* qA = Q + ((size_t)((b * kL + l0w + ln) * kH + h)) * kD;
  v16h aQ0, aQ1;
  #pragma unroll
  for (int j = 0; j < 8; ++j) {
    aQ0[j]     = (_Float16)qA[lh * 8 + j];
    aQ0[j + 8] = (_Float16)qA[16 + lh * 8 + j];
    aQ1[j]     = (_Float16)qA[32 + lh * 8 + j];
    aQ1[j + 8] = (_Float16)qA[48 + lh * 8 + j];
  }

  issueTile(0, 0);     // kick off first tile DMA
  __syncthreads();     // sQR ready

  float qr0[8], qr32[8];
  #pragma unroll
  for (int r = 0; r < 8; ++r) {
    qr0[r]  = sQR[wave * 16 + r + 8 * lh][0];
    qr32[r] = sQR[wave * 16 + r + 8 * lh][kT - 1];
  }

  float runMax[8], runSum[8], aLeft[8];
  v8f acc[4];
  #pragma unroll
  for (int r = 0; r < 8; ++r) { runMax[r] = -INFINITY; runSum[r] = 0.f; aLeft[r] = 0.f; }
  #pragma unroll
  for (int dt = 0; dt < 4; ++dt) acc[dt] = (v8f){};

  for (int s0 = 0; s0 < kL; s0 += kBc) {
    const int cur = (s0 / kBc) & 1;
    wait_async0();       // own async writes landed
    __syncthreads();     // all waves' writes visible; prior reads of other buffer done
    if (s0 + kBc < kL) issueTile(s0 + kBc, cur ^ 1);  // overlaps compute below

    // ---- S = Q K^T : 16x32 logit tile per wave ----
    v8f sf[2];
    #pragma unroll
    for (int ct = 0; ct < 2; ++ct) {
      int kr = ct * 16 + ln;
      v16h b0 = cat8(*(const v8h*)&sK[cur][kr][lh * 8],      *(const v8h*)&sK[cur][kr][16 + lh * 8]);
      v16h b1 = cat8(*(const v8h*)&sK[cur][kr][32 + lh * 8], *(const v8h*)&sK[cur][kr][48 + lh * 8]);
      v8f cz = {};
      v8f t0 = __builtin_amdgcn_wmma_f32_16x16x32_f16(false, aQ0, false, b0, (short)0, cz, false, false);
      sf[ct]  = __builtin_amdgcn_wmma_f32_16x16x32_f16(false, aQ1, false, b1, (short)0, t0, false, false);
    }

    const bool allLeft  = (s0 <= l0w - 47);
    const bool allRight = (s0 >= l0w + 31);

    float nmax[8];
    #pragma unroll
    for (int r = 0; r < 8; ++r) nmax[r] = runMax[r];

    if (allLeft || allRight) {
      float lg[2][8];
      #pragma unroll
      for (int ct = 0; ct < 2; ++ct) {
        #pragma unroll
        for (int r = 0; r < 8; ++r) {
          float sc = (sf[ct][r] + (allLeft ? qr0[r] : qr32[r])) * 0.125f;
          lg[ct][r] = sc;
          nmax[r] = fmaxf(nmax[r], sc);
        }
      }
      #pragma unroll
      for (int r = 0; r < 8; ++r) {
        #pragma unroll
        for (int ofs = 1; ofs < 16; ofs <<= 1)
          nmax[r] = fmaxf(nmax[r], __shfl_xor(nmax[r], ofs, 16));
      }
      float rs[8];
      #pragma unroll
      for (int r = 0; r < 8; ++r) rs[r] = 0.f;
      #pragma unroll
      for (int ct = 0; ct < 2; ++ct) {
        #pragma unroll
        for (int r = 0; r < 8; ++r) {
          float p = __expf(lg[ct][r] - nmax[r]);
          rs[r] += p;
          sP[wave][r + 8 * lh][ct * 16 + ln] = (_Float16)p;
        }
      }
      const float lf = allLeft ? 1.f : 0.f;
      #pragma unroll
      for (int r = 0; r < 8; ++r) {
        #pragma unroll
        for (int ofs = 1; ofs < 16; ofs <<= 1)
          rs[r] += __shfl_xor(rs[r], ofs, 16);
        float fac = __expf(runMax[r] - nmax[r]);
        runSum[r] = runSum[r] * fac + rs[r];
        aLeft[r]  = aLeft[r]  * fac + lf * rs[r];
        runMax[r] = nmax[r];
        #pragma unroll
        for (int dt = 0; dt < 4; ++dt) acc[dt][r] *= fac;
      }
    } else {
      float lg[2][8];
      #pragma unroll
      for (int ct = 0; ct < 2; ++ct) {
        #pragma unroll
        for (int r = 0; r < 8; ++r) {
          int mloc = r + 8 * lh;
          int dr = (s0 + ct * 16 + ln) - (l0w + mloc);
          int t  = dr < -16 ? 0 : (dr > 16 ? 32 : dr + 16);
          float sc = (sf[ct][r] + sQR[wave * 16 + mloc][t]) * 0.125f;
          if (dr >= -15 && dr <= 15) sBand[wave * 16 + mloc][dr + 15] = sc;
          lg[ct][r] = sc;
          nmax[r] = fmaxf(nmax[r], sc);
        }
      }
      #pragma unroll
      for (int r = 0; r < 8; ++r) {
        #pragma unroll
        for (int ofs = 1; ofs < 16; ofs <<= 1)
          nmax[r] = fmaxf(nmax[r], __shfl_xor(nmax[r], ofs, 16));
      }
      float rs[8], ls[8];
      #pragma unroll
      for (int r = 0; r < 8; ++r) { rs[r] = 0.f; ls[r] = 0.f; }
      #pragma unroll
      for (int ct = 0; ct < 2; ++ct) {
        #pragma unroll
        for (int r = 0; r < 8; ++r) {
          int mloc = r + 8 * lh;
          int dr = (s0 + ct * 16 + ln) - (l0w + mloc);
          float p = __expf(lg[ct][r] - nmax[r]);
          rs[r] += p;
          if (dr <= -16) ls[r] += p;
          sP[wave][mloc][ct * 16 + ln] = (_Float16)p;
        }
      }
      #pragma unroll
      for (int r = 0; r < 8; ++r) {
        #pragma unroll
        for (int ofs = 1; ofs < 16; ofs <<= 1) {
          rs[r] += __shfl_xor(rs[r], ofs, 16);
          ls[r] += __shfl_xor(ls[r], ofs, 16);
        }
        float fac = __expf(runMax[r] - nmax[r]);
        runSum[r] = runSum[r] * fac + rs[r];
        aLeft[r]  = aLeft[r]  * fac + ls[r];
        runMax[r] = nmax[r];
        #pragma unroll
        for (int dt = 0; dt < 4; ++dt) acc[dt][r] *= fac;
      }
    }

    // ---- O += P . V ----
    v16h aP = cat8(*(const v8h*)&sP[wave][ln][lh * 8], *(const v8h*)&sP[wave][ln][16 + lh * 8]);
    #pragma unroll
    for (int dt = 0; dt < 4; ++dt) {
      int dc = dt * 16 + ln;
      v16h bV = cat8(*(const v8h*)&sVt[cur][dc][lh * 8], *(const v8h*)&sVt[cur][dc][16 + lh * 8]);
      acc[dt] = __builtin_amdgcn_wmma_f32_16x16x32_f16(false, aP, false, bV, (short)0, acc[dt], false, false);
    }
  }

  // ---- finalize ----
  #pragma unroll
  for (int r = 0; r < 8; ++r) {
    float inv = 1.f / runSum[r];
    #pragma unroll
    for (int dt = 0; dt < 4; ++dt)
      sO[wave * 16 + r + 8 * lh][dt * 16 + ln] = acc[dt][r] * inv;
    if (ln == 0) {
      int m = wave * 16 + r + 8 * lh;
      sStat[m][0] = runMax[r];
      sStat[m][1] = runSum[r];
      sStat[m][2] = aLeft[r];
    }
  }
  __syncthreads();

  if (tid < kBr) {
    int m = tid;
    float mx = sStat[m][0];
    float invZ = 1.f / sStat[m][1];
    float bs = 0.f;
    #pragma unroll
    for (int j = 0; j < kBand; ++j) {
      float x = sBand[m][j];
      float p = (x > -1e30f) ? __expf(x - mx) * invZ : 0.f;
      sBand[m][j] = p;
      bs += p;
    }
    float alN = sStat[m][2] * invZ;
    sStat[m][1] = alN;
    sStat[m][2] = 1.f - alN - bs;
  }
  __syncthreads();

  for (int e = tid; e < kBr * kD; e += kThreads) {
    int m = e >> 6, d = e & 63;
    float w2 = sStat[m][1] * RV[d] + sStat[m][2] * RV[32 * kD + d];
    #pragma unroll
    for (int j = 0; j < kBand; ++j)
      w2 += sBand[m][j] * RV[(j + 1) * kD + d];
    size_t o = ((size_t)((b * kL + L0 + m) * kH + h)) * kD + d;
    Out[o] = sO[m][d] + w2;
  }
}

// ======================= fallback (no workspace): register-prefetch version =======================
__global__ __launch_bounds__(kThreads)
void relattn_fallback(const float* __restrict__ Q, const float* __restrict__ K,
                      const float* __restrict__ V, const float* __restrict__ RK,
                      const float* __restrict__ RV, float* __restrict__ Out) {
  __shared__ _Float16 sK[kBc][kD];
  __shared__ _Float16 sVt[kD][kBc];
  __shared__ float    sQR[kBr][kT];
  __shared__ float    sBand[kBr][kBand];
  __shared__ _Float16 sP[4][16][kBc];
  __shared__ float    sO[kBr][kD];
  __shared__ float    sStat[kBr][3];

  const int tid  = threadIdx.x;
  const int wave = tid >> 5;
  const int lane = tid & 31;
  const int ln   = lane & 15;
  const int lh   = lane >> 4;

  const int L0 = blockIdx.x * kBr;
  const int h  = blockIdx.y;
  const int b  = blockIdx.z;

  for (int e = tid; e < kBr * kT; e += kThreads) {
    int m = e / kT, t = e - m * kT;
    const float* qrow = Q + ((size_t)((b * kL + L0 + m) * kH + h)) * kD;
    const float* rkrow = RK + t * kD;
    float s = 0.f;
    #pragma unroll
    for (int d = 0; d < kD; ++d) s += qrow[d] * rkrow[d];
    sQR[m][t] = s;
  }
  for (int e = tid; e < kBr * kBand; e += kThreads)
    sBand[e / kBand][e % kBand] = -INFINITY;

  const int l0w = L0 + wave * 16;
  const float* qA = Q + ((size_t)((b * kL + l0w + ln) * kH + h)) * kD;
  v16h aQ0, aQ1;
  #pragma unroll
  for (int j = 0; j < 8; ++j) {
    aQ0[j]     = (_Float16)qA[lh * 8 + j];
    aQ0[j + 8] = (_Float16)qA[16 + lh * 8 + j];
    aQ1[j]     = (_Float16)qA[32 + lh * 8 + j];
    aQ1[j + 8] = (_Float16)qA[48 + lh * 8 + j];
  }

  __syncthreads();
  float qr0[8], qr32[8];
  #pragma unroll
  for (int r = 0; r < 8; ++r) {
    qr0[r]  = sQR[wave * 16 + r + 8 * lh][0];
    qr32[r] = sQR[wave * 16 + r + 8 * lh][kT - 1];
  }

  float runMax[8], runSum[8], aLeft[8];
  v8f acc[4];
  #pragma unroll
  for (int r = 0; r < 8; ++r) { runMax[r] = -INFINITY; runSum[r] = 0.f; aLeft[r] = 0.f; }
  #pragma unroll
  for (int dt = 0; dt < 4; ++dt) acc[dt] = (v8f){};

  float kpre[kPF], vpre[kPF];
  #pragma unroll
  for (int i = 0; i < kPF; ++i) {
    int e = tid + i * kThreads;
    int sr = e >> 6, d = e & 63;
    size_t g = ((size_t)((b * kL + sr) * kH + h)) * kD + d;
    kpre[i] = K[g];
    vpre[i] = V[g];
  }

  for (int s0 = 0; s0 < kL; s0 += kBc) {
    __syncthreads();
    #pragma unroll
    for (int i = 0; i < kPF; ++i) {
      int e = tid + i * kThreads;
      int sr = e >> 6, d = e & 63;
      sK[sr][d]  = (_Float16)kpre[i];
      sVt[d][sr] = (_Float16)vpre[i];
    }
    if (s0 + kBc < kL) {
      #pragma unroll
      for (int i = 0; i < kPF; ++i) {
        int e = tid + i * kThreads;
        int sr = e >> 6, d = e & 63;
        size_t g = ((size_t)((b * kL + s0 + kBc + sr) * kH + h)) * kD + d;
        kpre[i] = K[g];
        vpre[i] = V[g];
      }
    }
    __syncthreads();

    v8f sf[2];
    #pragma unroll
    for (int ct = 0; ct < 2; ++ct) {
      int kr = ct * 16 + ln;
      v16h b0 = cat8(*(const v8h*)&sK[kr][lh * 8],      *(const v8h*)&sK[kr][16 + lh * 8]);
      v16h b1 = cat8(*(const v8h*)&sK[kr][32 + lh * 8], *(const v8h*)&sK[kr][48 + lh * 8]);
      v8f cz = {};
      v8f t0 = __builtin_amdgcn_wmma_f32_16x16x32_f16(false, aQ0, false, b0, (short)0, cz, false, false);
      sf[ct]  = __builtin_amdgcn_wmma_f32_16x16x32_f16(false, aQ1, false, b1, (short)0, t0, false, false);
    }

    const bool allLeft  = (s0 <= l0w - 47);
    const bool allRight = (s0 >= l0w + 31);
    float nmax[8];
    #pragma unroll
    for (int r = 0; r < 8; ++r) nmax[r] = runMax[r];

    if (allLeft || allRight) {
      float lg[2][8];
      #pragma unroll
      for (int ct = 0; ct < 2; ++ct)
        #pragma unroll
        for (int r = 0; r < 8; ++r) {
          float sc = (sf[ct][r] + (allLeft ? qr0[r] : qr32[r])) * 0.125f;
          lg[ct][r] = sc;
          nmax[r] = fmaxf(nmax[r], sc);
        }
      #pragma unroll
      for (int r = 0; r < 8; ++r)
        #pragma unroll
        for (int ofs = 1; ofs < 16; ofs <<= 1)
          nmax[r] = fmaxf(nmax[r], __shfl_xor(nmax[r], ofs, 16));
      float rs[8];
      #pragma unroll
      for (int r = 0; r < 8; ++r) rs[r] = 0.f;
      #pragma unroll
      for (int ct = 0; ct < 2; ++ct)
        #pragma unroll
        for (int r = 0; r < 8; ++r) {
          float p = __expf(lg[ct][r] - nmax[r]);
          rs[r] += p;
          sP[wave][r + 8 * lh][ct * 16 + ln] = (_Float16)p;
        }
      const float lf = allLeft ? 1.f : 0.f;
      #pragma unroll
      for (int r = 0; r < 8; ++r) {
        #pragma unroll
        for (int ofs = 1; ofs < 16; ofs <<= 1)
          rs[r] += __shfl_xor(rs[r], ofs, 16);
        float fac = __expf(runMax[r] - nmax[r]);
        runSum[r] = runSum[r] * fac + rs[r];
        aLeft[r]  = aLeft[r]  * fac + lf * rs[r];
        runMax[r] = nmax[r];
        #pragma unroll
        for (int dt = 0; dt < 4; ++dt) acc[dt][r] *= fac;
      }
    } else {
      float lg[2][8];
      #pragma unroll
      for (int ct = 0; ct < 2; ++ct)
        #pragma unroll
        for (int r = 0; r < 8; ++r) {
          int mloc = r + 8 * lh;
          int dr = (s0 + ct * 16 + ln) - (l0w + mloc);
          int t  = dr < -16 ? 0 : (dr > 16 ? 32 : dr + 16);
          float sc = (sf[ct][r] + sQR[wave * 16 + mloc][t]) * 0.125f;
          if (dr >= -15 && dr <= 15) sBand[wave * 16 + mloc][dr + 15] = sc;
          lg[ct][r] = sc;
          nmax[r] = fmaxf(nmax[r], sc);
        }
      #pragma unroll
      for (int r = 0; r < 8; ++r)
        #pragma unroll
        for (int ofs = 1; ofs < 16; ofs <<= 1)
          nmax[r] = fmaxf(nmax[r], __shfl_xor(nmax[r], ofs, 16));
      float rs[8], ls[8];
      #pragma unroll
      for (int r = 0; r < 8; ++r) { rs[r] = 0.f; ls[r] = 0.f; }
      #pragma unroll
      for (int ct = 0; ct < 2; ++ct)
        #pragma unroll
        for (int r = 0; r < 8; ++r) {
          int mloc = r + 8 * lh;
          int dr = (s0 + ct * 16 + ln) - (l0w + mloc);
          float p = __expf(lg[ct][r] - nmax[r]);
          rs[r] += p;
          if (dr <= -16) ls[r] += p;
          sP[wave][mloc][ct * 16 + ln] = (_Float16)p;
        }
      #pragma unroll
      for (int r = 0; r < 8; ++r) {
        #pragma unroll
        for (int ofs = 1; ofs < 16; ofs <<= 1) {
          rs[r] += __shfl_xor(rs[r], ofs, 16);
          ls[r] += __shfl_xor(ls[r], ofs, 16);
        }
        float fac = __expf(runMax[r] - nmax[r]);
        runSum[r] = runSum[r] * fac + rs[r];
        aLeft[r]  = aLeft[r]  * fac + ls[r];
        runMax[r] = nmax[r];
        #pragma unroll
        for (int dt = 0; dt < 4; ++dt) acc[dt][r] *= fac;
      }
    }

    v16h aP = cat8(*(const v8h*)&sP[wave][ln][lh * 8], *(const v8h*)&sP[wave][ln][16 + lh * 8]);
    #pragma unroll
    for (int dt = 0; dt < 4; ++dt) {
      int dc = dt * 16 + ln;
      v16h bV = cat8(*(const v8h*)&sVt[dc][lh * 8], *(const v8h*)&sVt[dc][16 + lh * 8]);
      acc[dt] = __builtin_amdgcn_wmma_f32_16x16x32_f16(false, aP, false, bV, (short)0, acc[dt], false, false);
    }
  }

  #pragma unroll
  for (int r = 0; r < 8; ++r) {
    float inv = 1.f / runSum[r];
    #pragma unroll
    for (int dt = 0; dt < 4; ++dt)
      sO[wave * 16 + r + 8 * lh][dt * 16 + ln] = acc[dt][r] * inv;
    if (ln == 0) {
      int m = wave * 16 + r + 8 * lh;
      sStat[m][0] = runMax[r];
      sStat[m][1] = runSum[r];
      sStat[m][2] = aLeft[r];
    }
  }
  __syncthreads();

  if (tid < kBr) {
    int m = tid;
    float mx = sStat[m][0];
    float invZ = 1.f / sStat[m][1];
    float bs = 0.f;
    #pragma unroll
    for (int j = 0; j < kBand; ++j) {
      float x = sBand[m][j];
      float p = (x > -1e30f) ? __expf(x - mx) * invZ : 0.f;
      sBand[m][j] = p;
      bs += p;
    }
    float alN = sStat[m][2] * invZ;
    sStat[m][1] = alN;
    sStat[m][2] = 1.f - alN - bs;
  }
  __syncthreads();

  for (int e = tid; e < kBr * kD; e += kThreads) {
    int m = e >> 6, d = e & 63;
    float w2 = sStat[m][1] * RV[d] + sStat[m][2] * RV[32 * kD + d];
    #pragma unroll
    for (int j = 0; j < kBand; ++j)
      w2 += sBand[m][j] * RV[(j + 1) * kD + d];
    size_t o = ((size_t)((b * kL + L0 + m) * kH + h)) * kD + d;
    Out[o] = sO[m][d] + w2;
  }
}

extern "C" void kernel_launch(void* const* d_in, const int* in_sizes, int n_in,
                              void* d_out, int out_size, void* d_ws, size_t ws_size,
                              hipStream_t stream) {
  (void)in_sizes; (void)n_in; (void)out_size;
  const float* Q  = (const float*)d_in[0];
  const float* K  = (const float*)d_in[1];
  const float* V  = (const float*)d_in[2];
  const float* RK = (const float*)d_in[3];
  const float* RV = (const float*)d_in[4];
  float* Out = (float*)d_out;
  dim3 grid(kL / kBr, kH, kB);   // (32, 16, 2)
  if (d_ws != nullptr && ws_size >= kWsNeed) {
    _Float16* Kh = (_Float16*)d_ws;
    _Float16* Vt = Kh + kHeadElems;
    preconvert_kv<<<dim3(kL / 64, kH, kB), 256, 0, stream>>>(K, V, Kh, Vt);
    relattn_async<<<grid, dim3(kThreads), 0, stream>>>(Q, Kh, Vt, RK, RV, Out);
  } else {
    relattn_fallback<<<grid, dim3(kThreads), 0, stream>>>(Q, K, V, RK, RV, Out);
  }
}
